// MultiWaveletTransform_4123168604348
// MI455X (gfx1250) — compile-verified, matrix operand order checked by reading, not run on hardware
//
#include <hip/hip_runtime.h>
#include <hip/hip_bf16.h>

// MI455X / gfx1250: wave32, WMMA bf16 (f32 accumulate) for all heavy matmuls.
// float4 (b128) global loads on all contiguous streams; global_prefetch on the
// strided spectral-weight walk.

typedef __attribute__((ext_vector_type(16))) __bf16 v16bf;
typedef __attribute__((ext_vector_type(8)))  float  v8f;

namespace {
constexpr int kB     = 8;
constexpr int kSeq   = 2048;
constexpr int kC     = 128;
constexpr int kK     = 8;
constexpr int kCK    = 1024;   // C*K
constexpr int kICH   = 512;
constexpr int kAlpha = 16;
constexpr int kNS    = 11;     // floor(log2(2048))
}

static __device__ __forceinline__ __bf16 f2bf(float f) {
  unsigned int u = __builtin_bit_cast(unsigned int, f);
  unsigned int r = (u + 0x7FFFu + ((u >> 16) & 1u)) >> 16;   // round-to-nearest-even
  return __builtin_bit_cast(__bf16, (unsigned short)(r & 0xFFFFu));
}

// Fragment loader for K-contiguous, 16B-aligned memory.
// ISA 16-bit 16x32 layout: lanes 0-15 hold K {0..7, 16..23}; lanes 16-31 hold
// K {8..15, 24..31}. Caller passes kb = (lane>=16 ? 8 : 0).
static __device__ __forceinline__ v16bf load_frag_contig(const float* __restrict__ p, int kb) {
  const float4* qa = reinterpret_cast<const float4*>(p + kb);        // K kb..kb+7
  const float4* qb = reinterpret_cast<const float4*>(p + kb + 16);   // K kb+16..kb+23
  float4 a0 = qa[0], a1 = qa[1], b0 = qb[0], b1 = qb[1];
  v16bf r;
  r[0]  = f2bf(a0.x); r[1]  = f2bf(a0.y); r[2]  = f2bf(a0.z); r[3]  = f2bf(a0.w);
  r[4]  = f2bf(a1.x); r[5]  = f2bf(a1.y); r[6]  = f2bf(a1.z); r[7]  = f2bf(a1.w);
  r[8]  = f2bf(b0.x); r[9]  = f2bf(b0.y); r[10] = f2bf(b0.z); r[11] = f2bf(b0.w);
  r[12] = f2bf(b1.x); r[13] = f2bf(b1.y); r[14] = f2bf(b1.z); r[15] = f2bf(b1.w);
  return r;
}

// Weight fragment for spectral mix: w has layout (i, o, f) = (CK, CK, 16).
// Element (k=i_local, n=o): stride over i is CK*ALPHA floats (64KB) -> scalar loads.
static __device__ __forceinline__ v16bf load_frag_weight(const float* __restrict__ w,
                                                         int ic, int o, int f, int hi) {
  v16bf r;
  const float* p = w + (size_t)o * kAlpha + f;
#pragma unroll
  for (int e = 0; e < 8; ++e) {
    int i0 = ic + hi * 8 + e;
    r[e]     = f2bf(p[(size_t)i0 * (kCK * kAlpha)]);
    r[e + 8] = f2bf(p[(size_t)(i0 + 16) * (kCK * kAlpha)]);
  }
  return r;
}

// ---------------------------------------------------------------------------
// D[M x N] = A[M x Kd] * W^T + bias   (W stored row-major (N, Kd))
// Block = 256 threads = 8 waves; each wave owns one 16x16 tile.
// Grid: (M/16, N/128).
// ---------------------------------------------------------------------------
__global__ void __launch_bounds__(256) wmma_gemm_bf16(
    const float* __restrict__ A, const float* __restrict__ W,
    const float* __restrict__ bias, float* __restrict__ D, int Kd, int N) {
  const int wave  = threadIdx.x >> 5;
  const int lane  = threadIdx.x & 31;
  const int l15   = lane & 15;
  const int hi    = lane >> 4;
  const int tileM = blockIdx.x * 16;
  const int tileN = (blockIdx.y * 8 + wave) * 16;

  const float* arow = A + (size_t)(tileM + l15) * Kd;
  const float* wrow = W + (size_t)(tileN + l15) * Kd;

  v8f acc = {0.f, 0.f, 0.f, 0.f, 0.f, 0.f, 0.f, 0.f};
  for (int kc = 0; kc < Kd; kc += 32) {
    if (kc + 32 < Kd) {
      __builtin_prefetch(arow + kc + 32, 0, 1);
      __builtin_prefetch(wrow + kc + 32, 0, 1);
    }
    v16bf af = load_frag_contig(arow + kc, hi * 8);
    v16bf bf = load_frag_contig(wrow + kc, hi * 8);
    acc = __builtin_amdgcn_wmma_f32_16x16x32_bf16(false, af, false, bf,
                                                  (short)0, acc, false, false);
  }
  const int col = tileN + l15;
  const float bv = bias[col];
#pragma unroll
  for (int v = 0; v < 8; ++v)
    D[(size_t)(tileM + 8 * hi + v) * N + col] = acc[v] + bv;   // C/D layout: M=v+8*hi, N=l15
}

// ---------------------------------------------------------------------------
// Fused complex spectral mix for one frequency f and one 16-wide output tile:
//   Ud_spec = Sd * A + Ss * B ;  Us_spec = Sd * C      (complex, i->o over CK)
// A-tile rows 0-7 = real(batch b), rows 8-15 = imag(batch b).
// Two WMMAs (vs wr and wi) per weight yield all cross terms; recombine via
// shfl_xor(16) since M=b sits in lanes 0-15 and M=b+8 in lanes 16-31.
// Grid: (CK/128, m); block = 256 (8 waves, one output tile each).
// ---------------------------------------------------------------------------
__global__ void __launch_bounds__(256) wmma_spectral_mix(
    const float* __restrict__ SdR, const float* __restrict__ SdI,
    const float* __restrict__ SsR, const float* __restrict__ SsI,
    const float* __restrict__ Awr, const float* __restrict__ Awi,
    const float* __restrict__ Bwr, const float* __restrict__ Bwi,
    const float* __restrict__ Cwr, const float* __restrict__ Cwi,
    float* __restrict__ UdR, float* __restrict__ UdI,
    float* __restrict__ UsR, float* __restrict__ UsI) {
  const int wave  = threadIdx.x >> 5;
  const int lane  = threadIdx.x & 31;
  const int l15   = lane & 15;
  const int hi    = lane >> 4;
  const int f     = blockIdx.y;
  const int tileO = (blockIdx.x * 8 + wave) * 16;

  const float* apd = (l15 < 8) ? (SdR + (size_t)(l15 * 16 + f) * kCK)
                               : (SdI + (size_t)((l15 - 8) * 16 + f) * kCK);
  const float* aps = (l15 < 8) ? (SsR + (size_t)(l15 * 16 + f) * kCK)
                               : (SsI + (size_t)((l15 - 8) * 16 + f) * kCK);

  v8f ud_wr = {0.f,0.f,0.f,0.f,0.f,0.f,0.f,0.f};
  v8f ud_wi = ud_wr, us_wr = ud_wr, us_wi = ud_wr;

  const int oc = tileO + l15;
  const size_t wofs = (size_t)oc * kAlpha + f;
  for (int ic = 0; ic < kCK; ic += 32) {
    if (ic + 32 < kCK) {
      // Prefetch next i-chunk of the strided weight walk (one line per lane).
      size_t pofs = (size_t)(ic + 32 + hi * 8) * (kCK * kAlpha) + wofs;
      __builtin_prefetch(Awr + pofs, 0, 1);
      __builtin_prefetch(Awi + pofs, 0, 1);
      __builtin_prefetch(Bwr + pofs, 0, 1);
      __builtin_prefetch(Bwi + pofs, 0, 1);
      __builtin_prefetch(Cwr + pofs, 0, 1);
      __builtin_prefetch(Cwi + pofs, 0, 1);
    }
    v16bf ad = load_frag_contig(apd + ic, hi * 8);
    v16bf as = load_frag_contig(aps + ic, hi * 8);
    v16bf bAr = load_frag_weight(Awr, ic, oc, f, hi);
    v16bf bAi = load_frag_weight(Awi, ic, oc, f, hi);
    v16bf bBr = load_frag_weight(Bwr, ic, oc, f, hi);
    v16bf bBi = load_frag_weight(Bwi, ic, oc, f, hi);
    v16bf bCr = load_frag_weight(Cwr, ic, oc, f, hi);
    v16bf bCi = load_frag_weight(Cwi, ic, oc, f, hi);
    ud_wr = __builtin_amdgcn_wmma_f32_16x16x32_bf16(false, ad, false, bAr, (short)0, ud_wr, false, false);
    ud_wr = __builtin_amdgcn_wmma_f32_16x16x32_bf16(false, as, false, bBr, (short)0, ud_wr, false, false);
    ud_wi = __builtin_amdgcn_wmma_f32_16x16x32_bf16(false, ad, false, bAi, (short)0, ud_wi, false, false);
    ud_wi = __builtin_amdgcn_wmma_f32_16x16x32_bf16(false, as, false, bBi, (short)0, ud_wi, false, false);
    us_wr = __builtin_amdgcn_wmma_f32_16x16x32_bf16(false, ad, false, bCr, (short)0, us_wr, false, false);
    us_wi = __builtin_amdgcn_wmma_f32_16x16x32_bf16(false, ad, false, bCi, (short)0, us_wi, false, false);
  }

#pragma unroll
  for (int v = 0; v < 8; ++v) {
    float xr_wi = __shfl_xor(ud_wi[v], 16, 32);  // imag-row * wi
    float xr_wr = __shfl_xor(ud_wr[v], 16, 32);  // imag-row * wr
    float yr_wi = __shfl_xor(us_wi[v], 16, 32);
    float yr_wr = __shfl_xor(us_wr[v], 16, 32);
    if (hi == 0) {
      size_t idx = (size_t)(v * 16 + f) * kCK + oc;   // batch = v
      UdR[idx] = ud_wr[v] - xr_wi;   // Re: Fr*wr - Fi*wi
      UdI[idx] = ud_wi[v] + xr_wr;   // Im: Fr*wi + Fi*wr
      UsR[idx] = us_wr[v] - yr_wi;
      UsI[idx] = us_wi[v] + yr_wr;
    }
  }
}

// ---------------------------------------------------------------------------
// Truncated DFT: first m bins of rfft along t. X layout (b, n, CK).
// Grid: (CK/256, m, B).
// ---------------------------------------------------------------------------
__global__ void __launch_bounds__(256) dft_k(const float* __restrict__ X,
                                             float* __restrict__ SR,
                                             float* __restrict__ SI, int n) {
  const int ck = blockIdx.x * 256 + threadIdx.x;
  const int f  = blockIdx.y;
  const int b  = blockIdx.z;
  float phi = 6.283185307179586f * (float)f / (float)n;
  float cs, sn;
  __sincosf(phi, &sn, &cs);
  float cr = 1.f, si = 0.f, fr = 0.f, fi = 0.f;
  const float* xp = X + (size_t)b * n * kCK + ck;
  for (int t = 0; t < n; ++t) {
    float xv = xp[(size_t)t * kCK];
    fr += xv * cr;
    fi -= xv * si;
    float nr = cr * cs - si * sn;
    si = si * cs + cr * sn;
    cr = nr;
  }
  size_t idx = (size_t)(b * 16 + f) * kCK + ck;
  SR[idx] = fr;
  SI[idx] = fi;
}

// ---------------------------------------------------------------------------
// Truncated inverse rfft: only first m bins nonzero; bin weight 2 except
// DC and Nyquist. Grid: (CK/256, n, B).
// ---------------------------------------------------------------------------
__global__ void __launch_bounds__(256) idft_k(const float* __restrict__ SR,
                                              const float* __restrict__ SI,
                                              float* __restrict__ Y, int n, int m) {
  const int ck = blockIdx.x * 256 + threadIdx.x;
  const int t  = blockIdx.y;
  const int b  = blockIdx.z;
  float base = 6.283185307179586f * (float)t / (float)n;
  float cs, sn;
  __sincosf(base, &sn, &cs);
  float cr = 1.f, ci = 0.f, acc = 0.f;
  for (int f = 0; f < m; ++f) {
    float w = (f == 0 || 2 * f == n) ? 1.f : 2.f;
    size_t idx = (size_t)(b * 16 + f) * kCK + ck;
    acc += w * (SR[idx] * cr - SI[idx] * ci);
    float nr = cr * cs - ci * sn;
    ci = ci * cs + cr * sn;
    cr = nr;
  }
  Y[((size_t)b * n + t) * kCK + ck] = acc / (float)n;
}

// ---------------------------------------------------------------------------
// Decompose: xa = [x[::2], x[1::2]] (16-vec); s = xa@ec_s, d = xa@ec_d.
// ---------------------------------------------------------------------------
__global__ void __launch_bounds__(256) decompose_k(const float* __restrict__ X,
                                                   const float* __restrict__ ecs,
                                                   const float* __restrict__ ecd,
                                                   float* __restrict__ S,
                                                   float* __restrict__ D,
                                                   int h, int total) {
  __shared__ float fs[128], fd[128];
  if (threadIdx.x < 128) {
    fs[threadIdx.x] = ecs[threadIdx.x];
    fd[threadIdx.x] = ecd[threadIdx.x];
  }
  __syncthreads();
  int idx = blockIdx.x * 256 + threadIdx.x;
  if (idx >= total) return;
  int c = idx & (kC - 1);
  int t = (idx >> 7) % h;
  int b = idx / (kC * h);
  size_t be = ((size_t)b * (2 * h) + 2 * t) * kCK + c * kK;
  const float4* xe = reinterpret_cast<const float4*>(X + be);
  const float4* xo = reinterpret_cast<const float4*>(X + be + kCK);
  float4 e0 = xe[0], e1 = xe[1], o0 = xo[0], o1 = xo[1];
  float xa[16] = {e0.x, e0.y, e0.z, e0.w, e1.x, e1.y, e1.z, e1.w,
                  o0.x, o0.y, o0.z, o0.w, o1.x, o1.y, o1.z, o1.w};
  size_t bo = ((size_t)b * h + t) * kCK + c * kK;
#pragma unroll
  for (int k = 0; k < 8; ++k) {
    float s = 0.f, d = 0.f;
#pragma unroll
    for (int j = 0; j < 16; ++j) {
      s += xa[j] * fs[j * 8 + k];
      d += xa[j] * fd[j * 8 + k];
    }
    S[bo + k] = s;
    D[bo + k] = d;
  }
}

// ---------------------------------------------------------------------------
// Reconstruct: x += Us; xc = [x, Ud]; even = xc@rc_e, odd = xc@rc_o; interleave.
// ---------------------------------------------------------------------------
__global__ void __launch_bounds__(256) recombine_k(const float* __restrict__ Xin,
                                                   const float* __restrict__ Us,
                                                   const float* __restrict__ Ud,
                                                   const float* __restrict__ rce,
                                                   const float* __restrict__ rco,
                                                   float* __restrict__ Xout,
                                                   int h, int total) {
  __shared__ float fe[128], fo[128];
  if (threadIdx.x < 128) {
    fe[threadIdx.x] = rce[threadIdx.x];
    fo[threadIdx.x] = rco[threadIdx.x];
  }
  __syncthreads();
  int idx = blockIdx.x * 256 + threadIdx.x;
  if (idx >= total) return;
  int c = idx & (kC - 1);
  int t = (idx >> 7) % h;
  int b = idx / (kC * h);
  size_t bi = ((size_t)b * h + t) * kCK + c * kK;
  float xc[16];
#pragma unroll
  for (int j = 0; j < 8; ++j) {
    xc[j]     = Xin[bi + j] + Us[bi + j];
    xc[8 + j] = Ud[bi + j];
  }
  size_t bo = ((size_t)b * (2 * h) + 2 * t) * kCK + c * kK;
#pragma unroll
  for (int k = 0; k < 8; ++k) {
    float e = 0.f, o = 0.f;
#pragma unroll
    for (int j = 0; j < 16; ++j) {
      e += xc[j] * fe[j * 8 + k];
      o += xc[j] * fo[j * 8 + k];
    }
    Xout[bo + k]       = e;
    Xout[bo + kCK + k] = o;
  }
}

// x = x @ T0_W.T + T0_b on the length-1 coarsest level (in-place).
__global__ void __launch_bounds__(256) t0_k(float* __restrict__ X,
                                            const float* __restrict__ T0W,
                                            const float* __restrict__ T0b) {
  int idx = blockIdx.x * 256 + threadIdx.x;
  if (idx >= kB * kC) return;
  size_t base = (size_t)idx * kK;
  float xin[8];
#pragma unroll
  for (int j = 0; j < 8; ++j) xin[j] = X[base + j];
#pragma unroll
  for (int k = 0; k < 8; ++k) {
    float acc = T0b[k];
#pragma unroll
    for (int j = 0; j < 8; ++j) acc += xin[j] * T0W[k * 8 + j];
    X[base + k] = acc;
  }
}

extern "C" void kernel_launch(void* const* d_in, const int* in_sizes, int n_in,
                              void* d_out, int out_size, void* d_ws, size_t ws_size,
                              hipStream_t stream) {
  (void)in_sizes; (void)n_in; (void)out_size; (void)ws_size;
  const float* values = (const float*)d_in[2];
  const float* W0  = (const float*)d_in[4];
  const float* b0  = (const float*)d_in[5];
  const float* W1  = (const float*)d_in[6];
  const float* b1  = (const float*)d_in[7];
  const float* Awr = (const float*)d_in[8];
  const float* Awi = (const float*)d_in[9];
  const float* Bwr = (const float*)d_in[10];
  const float* Bwi = (const float*)d_in[11];
  const float* Cwr = (const float*)d_in[12];
  const float* Cwi = (const float*)d_in[13];
  const float* T0W = (const float*)d_in[14];
  const float* T0b = (const float*)d_in[15];
  const float* ecs = (const float*)d_in[16];
  const float* ecd = (const float*)d_in[17];
  const float* rce = (const float*)d_in[18];
  const float* rco = (const float*)d_in[19];
  float* out = (float*)d_out;

  // Workspace layout (~292 MiB of f32):
  float* ws = (float*)d_ws;
  size_t off = 0;
  float* Xa  = ws + off; off += (size_t)kB * kSeq * kCK;   // ping
  float* Xb  = ws + off; off += (size_t)kB * kSeq * kCK;   // pong
  float* Dt  = ws + off; off += (size_t)kB * 1024 * kCK;   // per-level detail temp
  float* UdB = ws + off; off += (size_t)kB * 2047 * kCK;   // all-level Ud
  float* UsB = ws + off; off += (size_t)kB * 2047 * kCK;   // all-level Us
  const size_t specN = (size_t)kB * 16 * kCK;
  float* SdR = ws + off; off += specN;
  float* SdI = ws + off; off += specN;
  float* SsR = ws + off; off += specN;
  float* SsI = ws + off; off += specN;
  float* UdSR = ws + off; off += specN;
  float* UdSI = ws + off; off += specN;
  float* UsSR = ws + off; off += specN;
  float* UsSI = ws + off; off += specN;

  const int MROWS = kB * kSeq;   // 16384

  // 1) x = values @ W0^T + b0    (16384 x 512) x (512 x 1024)
  wmma_gemm_bf16<<<dim3(MROWS / 16, kCK / 128), 256, 0, stream>>>(
      values, W0, b0, Xa, kICH, kCK);

  // 2) Forward wavelet decomposition (11 levels)
  float* Xcur = Xa;
  float* Xnxt = Xb;
  size_t cum = 0;
  for (int j = 0; j < kNS; ++j) {
    int n = kSeq >> j;       // current x length
    int h = n >> 1;          // s/d length
    int m = h / 2 + 1; if (m > kAlpha) m = kAlpha;
    int totalF = kB * h * kC;

    decompose_k<<<(totalF + 255) / 256, 256, 0, stream>>>(Xcur, ecs, ecd, Xnxt, Dt, h, totalF);

    dim3 gS(kCK / 256, m, kB);
    dft_k<<<gS, 256, 0, stream>>>(Xnxt, SsR, SsI, h);
    dft_k<<<gS, 256, 0, stream>>>(Dt,   SdR, SdI, h);

    wmma_spectral_mix<<<dim3(kCK / 128, m), 256, 0, stream>>>(
        SdR, SdI, SsR, SsI, Awr, Awi, Bwr, Bwi, Cwr, Cwi,
        UdSR, UdSI, UsSR, UsSI);

    float* UdL = UdB + cum * (size_t)kB * kCK;
    float* UsL = UsB + cum * (size_t)kB * kCK;
    dim3 gI(kCK / 256, h, kB);
    idft_k<<<gI, 256, 0, stream>>>(UdSR, UdSI, UdL, h, m);
    idft_k<<<gI, 256, 0, stream>>>(UsSR, UsSI, UsL, h, m);

    cum += (size_t)h;
    float* t = Xcur; Xcur = Xnxt; Xnxt = t;
  }

  // 3) Coarsest-level transform
  t0_k<<<(kB * kC + 255) / 256, 256, 0, stream>>>(Xcur, T0W, T0b);

  // 4) Reconstruction
  for (int i = kNS - 1; i >= 0; --i) {
    int h = 1024 >> i;                      // stored level length == current x length
    size_t lvlOff = (size_t)(2048 - 2 * h); // sum of previous level lengths
    float* UdL = UdB + lvlOff * (size_t)kB * kCK;
    float* UsL = UsB + lvlOff * (size_t)kB * kCK;
    int totalF = kB * h * kC;
    recombine_k<<<(totalF + 255) / 256, 256, 0, stream>>>(
        Xcur, UsL, UdL, rce, rco, Xnxt, h, totalF);
    float* t = Xcur; Xcur = Xnxt; Xnxt = t;
  }

  // 5) out = x @ W1^T + b1    (16384 x 1024) x (1024 x 512)
  wmma_gemm_bf16<<<dim3(MROWS / 16, kICH / 128), 256, 0, stream>>>(
      Xcur, W1, b1, out, kCK, kICH);
}
